// SlotAttention_52750788329585
// MI455X (gfx1250) — compile-verified, matrix operand order checked by reading, not run on hardware
//
#include <hip/hip_runtime.h>
#include <hip/hip_bf16.h>

// Slot Attention for MI455X (gfx1250, wave32, WMMA 16x16x32 f16->f32).
// All big contractions go through v_wmma_f32_16x16x32_f16; fp32 operands are
// converted to f16 in registers while building fragments (no f16 staging
// buffers). Input LayerNorm is fused into the projection GEMM's A loader.

#define EMB_D   768
#define NSLOT   8
#define KD      128
#define NITER   5
#define BATCH   128
#define NTOK    196
#define NROWTOK (BATCH * NTOK)      // 25088 (= 1568 * 16)
#define NSROW   (BATCH * NSLOT)     // 1024  (= 64 * 16)

typedef __attribute__((ext_vector_type(16))) _Float16 v16h;
typedef __attribute__((ext_vector_type(8)))  float    v8f;

// ---- Fragment loader: row-major fp32 matrix, one 16-value f16 fragment ----
// Works for A (lane = M row) and B (lane = N column, reading W[N,K] row-major).
// K chunking per ISA: halfwords 0..7 -> K = kb + hi*8 + j ; 8..15 -> +16.
__device__ inline v16h load_frag_f32(const float* __restrict__ base, int ld,
                                     int row, int kb, int hi) {
  const float* p = base + (size_t)row * ld + kb + hi * 8;
  v16h f;
#pragma unroll
  for (int j = 0; j < 8; ++j) f[j] = (_Float16)p[j];
#pragma unroll
  for (int j = 0; j < 8; ++j) f[8 + j] = (_Float16)p[16 + j];
  return f;
}

__device__ inline v8f wmma_step(v16h a, v16h b, v8f c) {
  return __builtin_amdgcn_wmma_f32_16x16x32_f16(false, a, false, b,
                                                (short)0, c, false, false);
}

// ---- per-token mean / rstd over EMB_D (for fused LN in projection) ----
__global__ void row_stats_kernel(const float* __restrict__ feats,
                                 float* __restrict__ stats) {
  __shared__ float s1[256], s2[256];
  int row = blockIdx.x, t = threadIdx.x;
  const float* p = feats + (size_t)row * EMB_D;
  float a0 = p[t], a1 = p[t + 256], a2 = p[t + 512];
  s1[t] = a0 + a1 + a2;
  s2[t] = a0 * a0 + a1 * a1 + a2 * a2;
  __syncthreads();
  for (int off = 128; off > 0; off >>= 1) {
    if (t < off) { s1[t] += s1[t + off]; s2[t] += s2[t + off]; }
    __syncthreads();
  }
  if (t == 0) {
    float mean = s1[0] / (float)EMB_D;
    float var  = s2[0] / (float)EMB_D - mean * mean;
    stats[2 * row]     = mean;
    stats[2 * row + 1] = rsqrtf(var + 1e-5f);
  }
}

// ---- fused LN(features) @ {Wk,Wv}^T  (WMMA)  grid(1568, 2) block(256) ----
__global__ void proj_ln_wmma_kernel(const float* __restrict__ feats,
                                    const float* __restrict__ stats,
                                    const float* __restrict__ g,
                                    const float* __restrict__ beta,
                                    const float* __restrict__ Wk,
                                    const float* __restrict__ Wv,
                                    float* __restrict__ kout,
                                    float* __restrict__ vout) {
  int lane = threadIdx.x & 31, wave = threadIdx.x >> 5;
  int l = lane & 15, hi = lane >> 4;
  int m0 = blockIdx.x * 16, n0 = wave * 16;
  int which = blockIdx.y;
  const float* W = which ? Wv : Wk;
  float scale = which ? 1.0f : 0.08838834764831845f;   // 128^-0.5 for K
  float* out = which ? vout : kout;

  int m = m0 + l;
  float mean = stats[2 * m], rstd = stats[2 * m + 1];
  const float* arow = feats + (size_t)m * EMB_D;
  v8f acc = {};
  for (int kb = 0; kb < EMB_D; kb += 32) {
    int c0 = kb + hi * 8;
    v16h a, b;
#pragma unroll
    for (int j = 0; j < 8; ++j) {
      int e0 = c0 + j, e1 = c0 + 16 + j;
      a[j]     = (_Float16)((arow[e0] - mean) * rstd * g[e0] + beta[e0]);
      a[8 + j] = (_Float16)((arow[e1] - mean) * rstd * g[e1] + beta[e1]);
    }
    const float* wrow = W + (size_t)(n0 + l) * EMB_D + c0;
#pragma unroll
    for (int j = 0; j < 8; ++j) { b[j] = (_Float16)wrow[j]; b[8 + j] = (_Float16)wrow[16 + j]; }
    acc = wmma_step(a, b, acc);
  }
  int n = n0 + l;
#pragma unroll
  for (int v = 0; v < 8; ++v)
    out[(size_t)(m0 + v + 8 * hi) * KD + n] = acc[v] * scale;
}

// ---- generic GEMM: out = act(A[M,K] @ W[N,K]^T + bias) (+residual) ----
// grid(M/16, N/16), block(32). K multiple of 32.
__global__ void gemm_wmma_kernel(const float* __restrict__ A, int lda,
                                 const float* __restrict__ W, int ldw,
                                 const float* __restrict__ bias,
                                 const float* __restrict__ residual,
                                 float* __restrict__ out, int ldo,
                                 int K, int do_relu) {
  int lane = threadIdx.x & 31;
  int l = lane & 15, hi = lane >> 4;
  int m0 = blockIdx.x * 16, n0 = blockIdx.y * 16;
  v8f acc = {};
  for (int kb = 0; kb < K; kb += 32) {
    v16h a = load_frag_f32(A, lda, m0 + l, kb, hi);
    v16h b = load_frag_f32(W, ldw, n0 + l, kb, hi);
    acc = wmma_step(a, b, acc);
  }
  int n = n0 + l;
  float bv = bias ? bias[n] : 0.f;
#pragma unroll
  for (int v = 0; v < 8; ++v) {
    int m = m0 + v + 8 * hi;
    float val = acc[v] + bv;
    if (do_relu) val = fmaxf(val, 0.f);
    if (residual) val += residual[(size_t)m * ldo + n];
    out[(size_t)m * ldo + n] = val;
  }
}

// ---- logits[b,n,k] = kproj[b,n,:] . q[b,k,:]   grid(B, 13) block(32) ----
__global__ void logits_wmma_kernel(const float* __restrict__ kproj,
                                   const float* __restrict__ q,
                                   float* __restrict__ logits) {
  int lane = threadIdx.x & 31;
  int l = lane & 15, hi = lane >> 4;
  int b = blockIdx.x, t0 = blockIdx.y * 16;
  const float* kb_ = kproj + (size_t)b * NTOK * KD;
  const float* qb  = q + (size_t)b * NSLOT * KD;
  v8f acc = {};
  for (int kb = 0; kb < KD; kb += 32) {
    int c0 = kb + hi * 8;
    v16h a = {}, bb = {};
    int tok = t0 + l;
    if (tok < NTOK) {
      const float* p = kb_ + (size_t)tok * KD + c0;
#pragma unroll
      for (int j = 0; j < 8; ++j) { a[j] = (_Float16)p[j]; a[8 + j] = (_Float16)p[16 + j]; }
    }
    if (l < NSLOT) {
      const float* p = qb + (size_t)l * KD + c0;
#pragma unroll
      for (int j = 0; j < 8; ++j) { bb[j] = (_Float16)p[j]; bb[8 + j] = (_Float16)p[16 + j]; }
    }
    acc = wmma_step(a, bb, acc);
  }
  if (l < NSLOT) {
#pragma unroll
    for (int v = 0; v < 8; ++v) {
      int tok = t0 + v + 8 * hi;
      if (tok < NTOK)
        logits[((size_t)b * NTOK + tok) * NSLOT + l] = acc[v];
    }
  }
}

// ---- updates[b,k,d] = sum_n attn_n[b,n,k] * v[b,n,d]  grid(B, 8) block(32) ----
__global__ void updates_wmma_kernel(const float* __restrict__ attn_n,
                                    const float* __restrict__ vproj,
                                    float* __restrict__ updates) {
  int lane = threadIdx.x & 31;
  int l = lane & 15, hi = lane >> 4;
  int b = blockIdx.x, d0 = blockIdx.y * 16;
  const float* an = attn_n + (size_t)b * NTOK * NSLOT;
  const float* vp = vproj + (size_t)b * NTOK * KD;
  int d = d0 + l;
  v8f acc = {};
  for (int kb = 0; kb < 224; kb += 32) {     // tokens padded 196 -> 224
    int c0 = kb + hi * 8;
    v16h a, bb;
#pragma unroll
    for (int j = 0; j < 8; ++j) {
      int t0_ = c0 + j, t1_ = c0 + 16 + j;
      float a0 = (l < NSLOT && t0_ < NTOK) ? an[(size_t)t0_ * NSLOT + l] : 0.f;
      float a1 = (l < NSLOT && t1_ < NTOK) ? an[(size_t)t1_ * NSLOT + l] : 0.f;
      a[j] = (_Float16)a0; a[8 + j] = (_Float16)a1;
      float b0 = (t0_ < NTOK) ? vp[(size_t)t0_ * KD + d] : 0.f;
      float b1 = (t1_ < NTOK) ? vp[(size_t)t1_ * KD + d] : 0.f;
      bb[j] = (_Float16)b0; bb[8 + j] = (_Float16)b1;
    }
    acc = wmma_step(a, bb, acc);
  }
#pragma unroll
  for (int v = 0; v < 8; ++v) {
    int slot = v + 8 * hi;
    if (slot < NSLOT)
      updates[((size_t)b * NSLOT + slot) * KD + d] = acc[v];
  }
}

// ---- LayerNorm over width 128, one row per block(128) ----
__global__ void ln128_kernel(const float* __restrict__ in,
                             const float* __restrict__ g,
                             const float* __restrict__ b,
                             float* __restrict__ out) {
  __shared__ float s1[128], s2[128];
  int row = blockIdx.x, t = threadIdx.x;
  float x = in[(size_t)row * KD + t];
  s1[t] = x; s2[t] = x * x;
  __syncthreads();
  for (int off = 64; off > 0; off >>= 1) {
    if (t < off) { s1[t] += s1[t + off]; s2[t] += s2[t + off]; }
    __syncthreads();
  }
  float mean = s1[0] / (float)KD;
  float var  = s2[0] / (float)KD - mean * mean;
  float r = rsqrtf(var + 1e-5f);
  out[(size_t)row * KD + t] = (x - mean) * r * g[t] + b[t];
}

__global__ void init_slots_kernel(const float* __restrict__ noise,
                                  const float* __restrict__ log_sigma,
                                  const float* __restrict__ mu,
                                  float* __restrict__ slots) {
  int idx = blockIdx.x * blockDim.x + threadIdx.x;
  if (idx >= NSROW * KD) return;
  int d = idx & (KD - 1);
  slots[idx] = noise[idx] * __expf(log_sigma[d]) + mu[d];
}

__global__ void softmax8_kernel(const float* __restrict__ logits,
                                float* __restrict__ attn_vis) {
  int idx = blockIdx.x * blockDim.x + threadIdx.x;   // b*196+n
  if (idx >= NROWTOK) return;
  const float* p = logits + (size_t)idx * NSLOT;
  float m = p[0];
#pragma unroll
  for (int k = 1; k < NSLOT; ++k) m = fmaxf(m, p[k]);
  float e[NSLOT], s = 0.f;
#pragma unroll
  for (int k = 0; k < NSLOT; ++k) { e[k] = __expf(p[k] - m); s += e[k]; }
  float inv = 1.f / s;
  float* o = attn_vis + (size_t)idx * NSLOT;
#pragma unroll
  for (int k = 0; k < NSLOT; ++k) o[k] = e[k] * inv;
}

__global__ void colsum_kernel(const float* __restrict__ vis,
                              float* __restrict__ colsum) {
  __shared__ float s[64];
  int bk = blockIdx.x, t = threadIdx.x;
  int b = bk >> 3, k = bk & 7;
  float acc = 0.f;
  for (int n = t; n < NTOK; n += 64)
    acc += vis[((size_t)b * NTOK + n) * NSLOT + k] + 1e-8f;
  s[t] = acc; __syncthreads();
  for (int off = 32; off > 0; off >>= 1) {
    if (t < off) s[t] += s[t + off];
    __syncthreads();
  }
  if (t == 0) colsum[bk] = s[0];
}

__global__ void attn_norm_kernel(const float* __restrict__ vis,
                                 const float* __restrict__ colsum,
                                 float* __restrict__ attn_n) {
  int idx = blockIdx.x * blockDim.x + threadIdx.x;
  if (idx >= NROWTOK * NSLOT) return;
  int k = idx & 7;
  int b = (idx >> 3) / NTOK;
  attn_n[idx] = (vis[idx] + 1e-8f) / colsum[b * NSLOT + k];
}

__global__ void gru_gate_kernel(const float* __restrict__ gi,
                                const float* __restrict__ gh,
                                const float* __restrict__ h_prev,
                                float* __restrict__ h_new) {
  int idx = blockIdx.x * blockDim.x + threadIdx.x;
  if (idx >= NSROW * KD) return;
  int row = idx >> 7, d = idx & (KD - 1);
  const float* pi = gi + (size_t)row * 3 * KD;
  const float* ph = gh + (size_t)row * 3 * KD;
  float r = 1.f / (1.f + __expf(-(pi[d] + ph[d])));
  float z = 1.f / (1.f + __expf(-(pi[KD + d] + ph[KD + d])));
  float n = tanhf(pi[2 * KD + d] + r * ph[2 * KD + d]);
  float h = h_prev[idx];
  h_new[idx] = (1.f - z) * n + z * h;
}

// ---- recon = sf_out(bkE) contracted with attn_vis over k; sum sq diff ----
__global__ void recon_loss_kernel(const float* __restrict__ sf_out,
                                  const float* __restrict__ attn_vis,
                                  const float* __restrict__ feats,
                                  float* __restrict__ loss_acc) {
  __shared__ float sred[256];
  __shared__ float a[NSLOT];
  int bn = blockIdx.x, t = threadIdx.x;
  int b = bn / NTOK;
  if (t < NSLOT) a[t] = attn_vis[(size_t)bn * NSLOT + t];
  __syncthreads();
  const float* sf = sf_out + (size_t)b * NSLOT * EMB_D;
  const float* f  = feats + (size_t)bn * EMB_D;
  float acc = 0.f;
  for (int e = t; e < EMB_D; e += 256) {
    float r = 0.f;
#pragma unroll
    for (int k = 0; k < NSLOT; ++k) r += sf[(size_t)k * EMB_D + e] * a[k];
    float d = r - f[e];
    acc += d * d;
  }
  sred[t] = acc; __syncthreads();
  for (int off = 128; off > 0; off >>= 1) {
    if (t < off) sred[t] += sred[t + off];
    __syncthreads();
  }
  if (t == 0) atomicAdd(loss_acc, sred[0]);
}

__global__ void finalize_loss_kernel(const float* __restrict__ loss_acc,
                                     float* __restrict__ out_loss) {
  out_loss[0] = loss_acc[0] * (1.f / ((float)NROWTOK * (float)EMB_D));
}

extern "C" void kernel_launch(void* const* d_in, const int* in_sizes, int n_in,
                              void* d_out, int out_size, void* d_ws, size_t ws_size,
                              hipStream_t stream) {
  const float* features  = (const float*)d_in[0];
  const float* slot_noise= (const float*)d_in[1];
  const float* ln_in_g   = (const float*)d_in[2];
  const float* ln_in_b   = (const float*)d_in[3];
  const float* ln_slot_g = (const float*)d_in[4];
  const float* ln_slot_b = (const float*)d_in[5];
  const float* ln_out_g  = (const float*)d_in[6];
  const float* ln_out_b  = (const float*)d_in[7];
  const float* ln_dec_g  = (const float*)d_in[8];
  const float* ln_dec_b  = (const float*)d_in[9];
  const float* mu        = (const float*)d_in[10];
  const float* log_sigma = (const float*)d_in[11];
  const float* Wk        = (const float*)d_in[12];
  const float* Wq        = (const float*)d_in[13];
  const float* Wv        = (const float*)d_in[14];
  const float* W_ih      = (const float*)d_in[15];
  const float* W_hh      = (const float*)d_in[16];
  const float* b_ih      = (const float*)d_in[17];
  const float* b_hh      = (const float*)d_in[18];
  const float* mlp_W1    = (const float*)d_in[19];
  const float* mlp_b1    = (const float*)d_in[20];
  const float* mlp_W2    = (const float*)d_in[21];
  const float* mlp_b2    = (const float*)d_in[22];
  const float* dec_W1    = (const float*)d_in[23];
  const float* dec_b1    = (const float*)d_in[24];
  const float* dec_W2    = (const float*)d_in[25];
  const float* dec_b2    = (const float*)d_in[26];
  const float* dec_W3    = (const float*)d_in[27];
  const float* dec_b3    = (const float*)d_in[28];

  float* ws = (float*)d_ws;
  size_t off = 0;
  auto alloc = [&](size_t n) { float* p = ws + off; off += n; return p; };
  float* stats     = alloc((size_t)NROWTOK * 2);
  float* kproj     = alloc((size_t)NROWTOK * KD);
  float* vproj     = alloc((size_t)NROWTOK * KD);
  float* slots     = alloc((size_t)NSROW * KD);
  float* s_ln      = alloc((size_t)NSROW * KD);
  float* q         = alloc((size_t)NSROW * KD);
  float* logits    = alloc((size_t)NROWTOK * NSLOT);
  float* attn_n    = alloc((size_t)NROWTOK * NSLOT);
  float* colsum    = alloc((size_t)BATCH * NSLOT);
  float* updates   = alloc((size_t)NSROW * KD);
  float* gi        = alloc((size_t)NSROW * 3 * KD);
  float* gh        = alloc((size_t)NSROW * 3 * KD);
  float* slots_gru = alloc((size_t)NSROW * KD);
  float* h_ln      = alloc((size_t)NSROW * KD);
  float* h1        = alloc((size_t)NSROW * KD);
  float* sf_ln     = alloc((size_t)NSROW * KD);
  float* h1d       = alloc((size_t)NSROW * 2 * KD);
  float* h2d       = alloc((size_t)NSROW * EMB_D);
  float* sf_out    = alloc((size_t)NSROW * EMB_D);
  float* loss_acc  = alloc(1);

  float* out       = (float*)d_out;
  float* out_slots = out;                                  // [128,8,128]
  float* out_attn  = out + (size_t)NSROW * KD;             // [128,196,8]
  float* out_loss  = out_attn + (size_t)NROWTOK * NSLOT;   // scalar

  // Stage 1: token stats + fused-LN projections (WMMA)
  row_stats_kernel<<<NROWTOK, 256, 0, stream>>>(features, stats);
  proj_ln_wmma_kernel<<<dim3(NROWTOK / 16, 2), 256, 0, stream>>>(
      features, stats, ln_in_g, ln_in_b, Wk, Wv, kproj, vproj);
  init_slots_kernel<<<(NSROW * KD + 255) / 256, 256, 0, stream>>>(
      slot_noise, log_sigma, mu, slots);

  // Stage 2: 5 slot-attention iterations
  for (int it = 0; it < NITER; ++it) {
    ln128_kernel<<<NSROW, 128, 0, stream>>>(slots, ln_slot_g, ln_slot_b, s_ln);
    gemm_wmma_kernel<<<dim3(NSROW / 16, KD / 16), 32, 0, stream>>>(
        s_ln, KD, Wq, KD, nullptr, nullptr, q, KD, KD, 0);
    logits_wmma_kernel<<<dim3(BATCH, 13), 32, 0, stream>>>(kproj, q, logits);
    softmax8_kernel<<<(NROWTOK + 255) / 256, 256, 0, stream>>>(logits, out_attn);
    colsum_kernel<<<BATCH * NSLOT, 64, 0, stream>>>(out_attn, colsum);
    attn_norm_kernel<<<(NROWTOK * NSLOT + 255) / 256, 256, 0, stream>>>(
        out_attn, colsum, attn_n);
    updates_wmma_kernel<<<dim3(BATCH, KD / 16), 32, 0, stream>>>(
        attn_n, vproj, updates);
    gemm_wmma_kernel<<<dim3(NSROW / 16, 3 * KD / 16), 32, 0, stream>>>(
        updates, KD, W_ih, KD, b_ih, nullptr, gi, 3 * KD, KD, 0);
    gemm_wmma_kernel<<<dim3(NSROW / 16, 3 * KD / 16), 32, 0, stream>>>(
        slots, KD, W_hh, KD, b_hh, nullptr, gh, 3 * KD, KD, 0);
    gru_gate_kernel<<<(NSROW * KD + 255) / 256, 256, 0, stream>>>(
        gi, gh, slots, slots_gru);
    ln128_kernel<<<NSROW, 128, 0, stream>>>(slots_gru, ln_out_g, ln_out_b, h_ln);
    gemm_wmma_kernel<<<dim3(NSROW / 16, KD / 16), 32, 0, stream>>>(
        h_ln, KD, mlp_W1, KD, mlp_b1, nullptr, h1, KD, KD, 1);
    gemm_wmma_kernel<<<dim3(NSROW / 16, KD / 16), 32, 0, stream>>>(
        h1, KD, mlp_W2, KD, mlp_b2, slots_gru, slots, KD, KD, 0);
  }
  hipMemcpyAsync(out_slots, slots, (size_t)NSROW * KD * sizeof(float),
                 hipMemcpyDeviceToDevice, stream);

  // Stage 3: decoder (WMMA chain) + reconstruction loss
  ln128_kernel<<<NSROW, 128, 0, stream>>>(slots, ln_dec_g, ln_dec_b, sf_ln);
  gemm_wmma_kernel<<<dim3(NSROW / 16, 2 * KD / 16), 32, 0, stream>>>(
      sf_ln, KD, dec_W1, KD, dec_b1, nullptr, h1d, 2 * KD, KD, 1);
  gemm_wmma_kernel<<<dim3(NSROW / 16, EMB_D / 16), 32, 0, stream>>>(
      h1d, 2 * KD, dec_W2, 2 * KD, dec_b2, nullptr, h2d, EMB_D, 2 * KD, 1);
  gemm_wmma_kernel<<<dim3(NSROW / 16, EMB_D / 16), 32, 0, stream>>>(
      h2d, EMB_D, dec_W3, EMB_D, dec_b3, nullptr, sf_out, EMB_D, EMB_D, 0);

  hipMemsetAsync(loss_acc, 0, sizeof(float), stream);
  recon_loss_kernel<<<NROWTOK, 256, 0, stream>>>(sf_out, out_attn, features, loss_acc);
  finalize_loss_kernel<<<1, 1, 0, stream>>>(loss_acc, out_loss);
}
